// HAttention1D_27419071218115
// MI455X (gfx1250) — matrix operand order repaired and checked by reading, not compiled, not run on hardware
//
#include <hip/hip_runtime.h>
#include <hip/hip_bf16.h>

// ---------------------------------------------------------------------------
// HAttention1D for MI455X (gfx1250, wave32, WMMA).
// B=4, N=4096, DIM=1024, HEADS=8, DH=64, BSZ=16, INNER=512, BH=32
// Levels 0..7; coarse tokens per bh = 4064; total tokens per bh = 8160.
// ---------------------------------------------------------------------------

typedef __attribute__((ext_vector_type(2)))  float  v2f;
typedef __attribute__((ext_vector_type(8)))  float  v8f;
typedef __attribute__((ext_vector_type(8)))  __bf16 v8bf;
typedef __attribute__((ext_vector_type(16))) __bf16 v16bf;

#if __has_builtin(__builtin_amdgcn_wmma_f32_16x16x4_f32)
#define HA_HAS_WMMA_F32 1
#else
#define HA_HAS_WMMA_F32 0
#endif

// D = A(16x4) * B(4x16) + C, fp32 WMMA (ISA layouts: A lane m=lane%16,
// k = (lane<16?0:2)+{0,1}; B symmetric with n=lane%16; C/D lane=n, vgpr=m(+8)).
__device__ __forceinline__ v8f ha_wmma_f32_16x16x4(v2f a, v2f b, v8f c) {
#if HA_HAS_WMMA_F32
  return __builtin_amdgcn_wmma_f32_16x16x4_f32(
      false, a, false, b, (short)0, c, false, false);
#else
  v8f d = c;
  const int lane = (int)(threadIdx.x & 31);
  const int lr = lane & 15, hi = lane >> 4;
#pragma unroll
  for (int k = 0; k < 4; ++k) {
    float av = (k & 1) ? a.y : a.x;   // valid on lanes with (lane>>4)==(k>>1)
    float bv = (k & 1) ? b.y : b.x;
    int grp = (k >> 1) << 4;
    float bn = __shfl(bv, lr + grp, 32);
#pragma unroll
    for (int rr = 0; rr < 8; ++rr) {
      float am = __shfl(av, (rr + hi * 8) + grp, 32);
      d[rr] += am * bn;
    }
  }
  return d;
#endif
}

__device__ __forceinline__ v16bf ha_cat8(v8bf lo, v8bf hi) {
  v16bf r;
#pragma unroll
  for (int i = 0; i < 8; ++i) { r[i] = lo[i]; r[i + 8] = hi[i]; }
  return r;
}

// ---------------------------------------------------------------------------
// Shared 128x128 bf16 WMMA GEMM core: C[128,128] += A[tileM:,:K] * B[tileN:,:K]^T
// A,B row-major with K contiguous. 8 waves, each wave owns 32(M) x 64(N).
// LDS rows padded to 40 bf16 (80 B) for bank-conflict-free fragment reads.
// ---------------------------------------------------------------------------
template <int KD>
__device__ __forceinline__ void ha_mma_128x128(
    const __bf16* __restrict__ A, const __bf16* __restrict__ Bm,
    int tileM, int tileN, __bf16* As, __bf16* Bs, v8f acc[2][4]) {
  const int tid  = (int)threadIdx.x;
  const int lane = tid & 31;
  const int w    = tid >> 5;
  const int wm   = (w >> 1) * 32;
  const int wn   = (w & 1) * 64;
  const int lr   = lane & 15;
  const int hi   = lane >> 4;
  const int r    = tid >> 1;          // 0..127 tile row loaded by this thread
  const int c    = (tid & 1) * 16;    // 0 or 16 (bf16 elements)

  for (int k0 = 0; k0 < KD; k0 += 32) {
    const uint4* ga = (const uint4*)(A + (size_t)(tileM + r) * KD + k0 + c);
    uint4 a0 = ga[0], a1 = ga[1];
    const uint4* gb = (const uint4*)(Bm + (size_t)(tileN + r) * KD + k0 + c);
    uint4 b0 = gb[0], b1 = gb[1];
    if (k0 + 32 < KD) {   // global_prefetch_b8 for next K tile
      __builtin_prefetch(A + (size_t)(tileM + r) * KD + k0 + 32 + c, 0, 1);
      __builtin_prefetch(Bm + (size_t)(tileN + r) * KD + k0 + 32 + c, 0, 1);
    }
    *(uint4*)&As[r * 40 + c]     = a0;
    *(uint4*)&As[r * 40 + c + 8] = a1;
    *(uint4*)&Bs[r * 40 + c]     = b0;
    *(uint4*)&Bs[r * 40 + c + 8] = b1;
    __syncthreads();

    // A frag (16x32 bf16): lane m=lr; k-base 0/8 per half-wave, elems
    // 0-7 -> k=base..base+7, 8-15 -> k=base+16..base+23.
    v16bf af[2];
    const int akb = hi ? 8 : 0;
#pragma unroll
    for (int mi = 0; mi < 2; ++mi) {
      const __bf16* ap = &As[(wm + mi * 16 + lr) * 40 + akb];
      af[mi] = ha_cat8(*(const v8bf*)ap, *(const v8bf*)(ap + 16));
    }
    // B frag (32x16 bf16): lane n=lr; same k order as A frag.
    v16bf bf[4];
    const int bkb = hi ? 8 : 0;
#pragma unroll
    for (int ni = 0; ni < 4; ++ni) {
      const __bf16* bp = &Bs[(wn + ni * 16 + lr) * 40 + bkb];
      bf[ni] = ha_cat8(*(const v8bf*)bp, *(const v8bf*)(bp + 16));
    }
#pragma unroll
    for (int mi = 0; mi < 2; ++mi)
#pragma unroll
      for (int ni = 0; ni < 4; ++ni)
        acc[mi][ni] = __builtin_amdgcn_wmma_f32_16x16x32_bf16(
            false, af[mi], false, bf[ni], (short)0, acc[mi][ni], false, false);
    __syncthreads();
  }
}

// ---------------------------------------------------------------------------
__global__ void __launch_bounds__(256) ha_f32_to_bf16(
    const float* __restrict__ src, __bf16* __restrict__ dst, int n) {
  int i = (int)(blockIdx.x * 256 + threadIdx.x);
  if (i < n) dst[i] = (__bf16)src[i];
}

// qkv = Xb(16384x1024) * Wq(1536x1024)^T ; scatter into Q/K/V [bh][8160][64]
__global__ void __launch_bounds__(256) ha_gemm_qkv(
    const __bf16* __restrict__ Xb, const __bf16* __restrict__ Wq,
    float* __restrict__ Q, float* __restrict__ K, float* __restrict__ V) {
  __shared__ __align__(16) __bf16 As[128 * 40];
  __shared__ __align__(16) __bf16 Bs[128 * 40];
  const int tileN = blockIdx.x * 128;   // 0..1408
  const int tileM = blockIdx.y * 128;   // 0..16256
  v8f acc[2][4];
  v8f z = {0.f, 0.f, 0.f, 0.f, 0.f, 0.f, 0.f, 0.f};
#pragma unroll
  for (int i = 0; i < 2; ++i)
#pragma unroll
    for (int j = 0; j < 4; ++j) acc[i][j] = z;

  ha_mma_128x128<1024>(Xb, Wq, tileM, tileN, As, Bs, acc);

  const int lane = (int)(threadIdx.x & 31), w = (int)(threadIdx.x >> 5);
  const int wm = (w >> 1) * 32, wn = (w & 1) * 64;
  const int lr = lane & 15, hi = lane >> 4;
#pragma unroll
  for (int mi = 0; mi < 2; ++mi)
#pragma unroll
    for (int ni = 0; ni < 4; ++ni)
#pragma unroll
      for (int rr = 0; rr < 8; ++rr) {
        int row = tileM + wm + mi * 16 + rr + hi * 8;   // 0..16383
        int col = tileN + wn + ni * 16 + lr;            // 0..1535
        float vv = acc[mi][ni][rr];
        int part = col >> 9;          // 0:q 1:k 2:v
        int cc = col & 511;
        int h = cc >> 6, d = cc & 63;
        int bb = row >> 12, nn = row & 4095;
        size_t dst = ((size_t)(bb * 8 + h) * 8160 + nn) * 64 + d;
        if (part == 0)      Q[dst] = vv * 0.125f;  // DIM_HEAD^-0.5 fused
        else if (part == 1) K[dst] = vv;
        else                V[dst] = vv;
      }
}

// Build levels 1..7 directly from level 0 (mean for q/k, sum for v).
__global__ void __launch_bounds__(256) ha_pyramid(
    float* __restrict__ Q, float* __restrict__ K, float* __restrict__ V) {
  int idx = (int)(blockIdx.x * 256 + threadIdx.x);  // < 32*4064*64
  int d = idx & 63;
  int rest = idx >> 6;
  int pos = rest % 4064;
  int bh = rest / 4064;
  int lvl = 1, cnt = 2048, loc = pos, coff = 0;
  while (loc >= cnt) { loc -= cnt; coff += cnt; cnt >>= 1; ++lvl; }
  int span = 1 << lvl;
  float inv = 1.0f / (float)span;
  size_t src = ((size_t)bh * 8160 + (size_t)loc * span) * 64 + d;
  float qa = 0.f, ka = 0.f, va = 0.f;
  for (int j = 0; j < span; ++j) {
    qa += Q[src + (size_t)j * 64];
    ka += K[src + (size_t)j * 64];
    va += V[src + (size_t)j * 64];
  }
  size_t dst = ((size_t)bh * 8160 + 4096 + coff + loc) * 64 + d;
  Q[dst] = qa * inv;
  K[dst] = ka * inv;
  V[dst] = va;
}

// One wave per (bh, level, block): S=q*k^T (fp32 WMMA, K=64), A=exp(S),
// rowsum(A), Y=A*v (fp32 WMMA, 16x64). Coarse levels use sibling key block.
__global__ void __launch_bounds__(256) ha_attention(
    const float* __restrict__ Q, const float* __restrict__ Kt,
    const float* __restrict__ V, float* __restrict__ Y,
    float* __restrict__ Asum) {
  __shared__ float Ash[8][16][17];
  const int lane = (int)(threadIdx.x & 31);
  const int w    = (int)(threadIdx.x >> 5);
  const int gw   = (int)blockIdx.x * 8 + w;     // 0..16319
  const int bh   = gw / 510;
  int blk = gw - bh * 510;
  int lvl = 0, nb = 256, tok = 0;
  while (blk >= nb) { blk -= nb; tok += nb * 16; nb >>= 1; ++lvl; }
  const int kblk = lvl ? (blk ^ 1) : blk;       // sibling flip on coarse levels
  const float* qb = Q  + ((size_t)bh * 8160 + tok + blk  * 16) * 64;
  const float* kb = Kt + ((size_t)bh * 8160 + tok + kblk * 16) * 64;
  const float* vb = V  + ((size_t)bh * 8160 + tok + blk  * 16) * 64;
  const int lr = lane & 15, hi = lane >> 4;

  // S = qb (16x64) * kb^T (64x16): 16 steps of K=4.
  v8f s = {0.f, 0.f, 0.f, 0.f, 0.f, 0.f, 0.f, 0.f};
#pragma unroll
  for (int k0 = 0; k0 < 64; k0 += 4) {
    int kk = k0 + hi * 2;
    v2f a; a.x = qb[lr * 64 + kk]; a.y = qb[lr * 64 + kk + 1];
    v2f b; b.x = kb[lr * 64 + kk]; b.y = kb[lr * 64 + kk + 1];
    s = ha_wmma_f32_16x16x4(a, b, s);
  }
  // A = exp(S), staged to LDS to convert C-layout -> A-fragment layout.
#pragma unroll
  for (int rr = 0; rr < 8; ++rr)
    Ash[w][rr + hi * 8][lr] = expf(s[rr]);
  __builtin_amdgcn_wave_barrier();   // wave-private LDS; DS ops are in-order

  if (hi == 0) {                     // lanes 0..15: row sums
    float rs = 0.f;
#pragma unroll
    for (int j = 0; j < 16; ++j) rs += Ash[w][lr][j];
    Asum[(size_t)bh * 8160 + tok + blk * 16 + lr] = rs;
  }

  // Y = A (16x16) * vb (16x64): 4 K-chunks x 4 N-tiles of fp32 WMMA.
  v8f y[4];
  v8f z = {0.f, 0.f, 0.f, 0.f, 0.f, 0.f, 0.f, 0.f};
#pragma unroll
  for (int t = 0; t < 4; ++t) y[t] = z;
#pragma unroll
  for (int c = 0; c < 4; ++c) {
    v2f a;
    a.x = Ash[w][lr][c * 4 + hi * 2];
    a.y = Ash[w][lr][c * 4 + hi * 2 + 1];
#pragma unroll
    for (int t = 0; t < 4; ++t) {
      v2f b;
      b.x = vb[(size_t)(c * 4 + hi * 2) * 64 + t * 16 + lr];
      b.y = vb[(size_t)(c * 4 + hi * 2 + 1) * 64 + t * 16 + lr];
      y[t] = ha_wmma_f32_16x16x4(a, b, y[t]);
    }
  }
  size_t ybase = ((size_t)bh * 8160 + tok + blk * 16) * 64;
#pragma unroll
  for (int t = 0; t < 4; ++t)
#pragma unroll
    for (int rr = 0; rr < 8; ++rr)
      Y[ybase + (size_t)(rr + hi * 8) * 64 + t * 16 + lr] = y[t][rr];
}

// out(n) = sum_l Y_l[n>>l] / (sum_l A_l[n>>l] + eps); emit bf16 O in (b,n,h*d).
__global__ void __launch_bounds__(256) ha_combine(
    const float* __restrict__ Y, const float* __restrict__ Asum,
    __bf16* __restrict__ O) {
  int idx = (int)(blockIdx.x * 256 + threadIdx.x);  // < 32*4096*64
  int d = idx & 63;
  int rest = idx >> 6;
  int n = rest & 4095;
  int bh = rest >> 12;
  const int offs[8] = {0, 4096, 6144, 7168, 7680, 7936, 8064, 8128};
  float ys = 0.f, as = 0.f;
#pragma unroll
  for (int l = 0; l < 8; ++l) {
    size_t base = (size_t)bh * 8160 + offs[l] + (n >> l);
    ys += Y[base * 64 + d];
    as += Asum[base];
  }
  float o = ys / (as + 1e-8f);
  int bb = bh >> 3, h = bh & 7;
  O[((size_t)(bb * 4096 + n)) * 512 + h * 64 + d] = (__bf16)o;
}

// out = Ob(16384x512) * Wo(1024x512)^T + bias ; fp32 result.
__global__ void __launch_bounds__(256) ha_gemm_out(
    const __bf16* __restrict__ Ob, const __bf16* __restrict__ Wo,
    const float* __restrict__ bias, float* __restrict__ out) {
  __shared__ __align__(16) __bf16 As[128 * 40];
  __shared__ __align__(16) __bf16 Bs[128 * 40];
  const int tileN = blockIdx.x * 128;   // 0..896
  const int tileM = blockIdx.y * 128;   // 0..16256
  v8f acc[2][4];
  v8f z = {0.f, 0.f, 0.f, 0.f, 0.f, 0.f, 0.f, 0.f};
#pragma unroll
  for (int i = 0; i < 2; ++i)
#pragma unroll
    for (int j = 0; j < 4; ++j) acc[i][j] = z;

  ha_mma_128x128<512>(Ob, Wo, tileM, tileN, As, Bs, acc);

  const int lane = (int)(threadIdx.x & 31), w = (int)(threadIdx.x >> 5);
  const int wm = (w >> 1) * 32, wn = (w & 1) * 64;
  const int lr = lane & 15, hi = lane >> 4;
#pragma unroll
  for (int mi = 0; mi < 2; ++mi)
#pragma unroll
    for (int ni = 0; ni < 4; ++ni)
#pragma unroll
      for (int rr = 0; rr < 8; ++rr) {
        int row = tileM + wm + mi * 16 + rr + hi * 8;
        int col = tileN + wn + ni * 16 + lr;
        out[(size_t)row * 1024 + col] = acc[mi][ni][rr] + bias[col];
      }
}

// ---------------------------------------------------------------------------
// Workspace layout (bytes, all 256-aligned):
//   Xb  @ 0          : 16384*1024 bf16  = 33554432
//   Wq  @ 33554432   : 1536*1024  bf16  =  3145728
//   Wo  @ 36700160   : 1024*512   bf16  =  1048576
//   Q   @ 37748736   : 32*8160*64 f32   = 66846720
//   K   @ 104595456  : 32*8160*64 f32   = 66846720
//   V   @ 171442176  : 32*8160*64 f32   = 66846720
//   Y   @ 238288896  : 32*8160*64 f32   = 66846720
//   A   @ 305135616  : 32*8160    f32   =  1044480
//   O   @ 306180096  : 16384*512  bf16  = 16777216   (total ~308 MB)
// ---------------------------------------------------------------------------
extern "C" void kernel_launch(void* const* d_in, const int* in_sizes, int n_in,
                              void* d_out, int out_size, void* d_ws, size_t ws_size,
                              hipStream_t stream) {
  const float* x     = (const float*)d_in[0];
  const float* w_qkv = (const float*)d_in[1];
  const float* w_out = (const float*)d_in[2];
  const float* b_out = (const float*)d_in[3];
  float* out = (float*)d_out;

  char* ws = (char*)d_ws;
  __bf16* Xb = (__bf16*)(ws + 0);
  __bf16* Wq = (__bf16*)(ws + 33554432);
  __bf16* Wo = (__bf16*)(ws + 36700160);
  float*  Q  = (float*)(ws + 37748736);
  float*  K  = (float*)(ws + 104595456);
  float*  V  = (float*)(ws + 171442176);
  float*  Y  = (float*)(ws + 238288896);
  float*  A  = (float*)(ws + 305135616);
  __bf16* O  = (__bf16*)(ws + 306180096);

  ha_f32_to_bf16<<<65536, 256, 0, stream>>>(x, Xb, 16777216);
  ha_f32_to_bf16<<<6144,  256, 0, stream>>>(w_qkv, Wq, 1572864);
  ha_f32_to_bf16<<<2048,  256, 0, stream>>>(w_out, Wo, 524288);

  ha_gemm_qkv<<<dim3(12, 128), 256, 0, stream>>>(Xb, Wq, Q, K, V);
  ha_pyramid<<<32512, 256, 0, stream>>>(Q, K, V);
  ha_attention<<<2040, 256, 0, stream>>>(Q, K, V, Y, A);
  ha_combine<<<32768, 256, 0, stream>>>(Y, A, O);
  ha_gemm_out<<<dim3(8, 128), 256, 0, stream>>>(O, Wo, b_out, out);
}
